// FTTransformerGINeFusedLayer_83777632076518
// MI455X (gfx1250) — matrix operand order repaired and checked by reading, not compiled, not run on hardware
//
#include <hip/hip_runtime.h>
#include <hip/hip_bf16.h>
#include <math.h>

#define C_DIM   256
#define NHEAD   8
#define HID     128
#define B_DIM   1024
#define S_DIM   32
#define N_NODES 50000
#define E_EDGES 640000
#define FD      384
#define HD      32      // C / NH
#define EPSV    1e-5f

typedef __attribute__((ext_vector_type(16))) __bf16 v16bf;
typedef __attribute__((ext_vector_type(8)))  float  v8f;

union FragB {
    v16bf v;
    struct { int4 lo; int4 hi; } i;
};

// ---------------------------------------------------------------- utilities

__global__ void cvt_bf16_kernel(const float* __restrict__ src,
                                __bf16* __restrict__ dst, int n) {
    int i = blockIdx.x * blockDim.x + threadIdx.x;
    if (i < n) dst[i] = (__bf16)src[i];
}

__global__ void zero_kernel(float* __restrict__ p, size_t n) {
    size_t i = (size_t)blockIdx.x * blockDim.x + threadIdx.x;
    if (i < n) p[i] = 0.0f;
}

__global__ void add_kernel(const float* __restrict__ a, const float* __restrict__ b,
                           float* __restrict__ o, size_t n) {
    size_t i = (size_t)blockIdx.x * blockDim.x + threadIdx.x;
    if (i < n) o[i] = a[i] + b[i];
}

// ---------------------------------------------------------------- WMMA GEMM
// C[M,N] = act(A[M,K] @ Bw[K,N] + bias)   (A f32 -> bf16 during LDS staging;
// Bw pre-converted bf16).  Optional fused edge epilogue:
//   val = relu(acc + bias[n] + gatherTab[src[m], n]); atomicAdd(scatterOut[dst[m], n], val)
// Workgroup: 256 threads = 8 waves (4x2); tile 128(M) x 64(N); wave tile 32x32
// (4 accumulators, 4 WMMAs per K-step with fragment reuse); K step 32.
__launch_bounds__(256)
__global__ void gemm_wmma_kernel(const float* __restrict__ A,
                                 const __bf16* __restrict__ Bw,
                                 const float* __restrict__ bias,
                                 float* __restrict__ Cout,
                                 int M, int N, int K, int relu,
                                 const int* __restrict__ srcIdx,
                                 const int* __restrict__ dstIdx,
                                 const float* __restrict__ gatherTab,
                                 float* __restrict__ scatterOut) {
    __shared__ __align__(16) __bf16 ldsA[128 * 48];  // 12 KB, row stride 48 bf16 (96B)
    __shared__ __align__(16) __bf16 ldsB[64 * 48];   //  6 KB, stored transposed: [n][k]

    const int tid  = threadIdx.x;
    const int lane = tid & 31;
    const int wave = tid >> 5;
    const int wm   = wave >> 1;        // 0..3 : 32-row band
    const int wn   = wave & 1;         // 0..1 : 32-col band
    const int hl   = lane >> 4;        // half-wave select
    const int l    = lane & 15;
    const int tileM = blockIdx.x * 128;
    const int tileN = blockIdx.y * 64;

    v8f acc[4];                        // [mi*2+ni]
    #pragma unroll
    for (int i = 0; i < 4; ++i)
        acc[i] = (v8f){0.f, 0.f, 0.f, 0.f, 0.f, 0.f, 0.f, 0.f};

    for (int k0 = 0; k0 < K; k0 += 32) {
        __syncthreads();
        // stage A tile: 128 x 32 f32 -> bf16, 1024 float4 chunks (4 per thread)
        #pragma unroll
        for (int i = 0; i < 4; ++i) {
            int chunk = i * 256 + tid;          // 0..1023
            int row = chunk >> 3;               // 8 float4 per row
            int c4  = chunk & 7;
            int gr  = tileM + row;
            float4 av = make_float4(0.f, 0.f, 0.f, 0.f);
            if (gr < M)
                av = *reinterpret_cast<const float4*>(A + (size_t)gr * K + k0 + c4 * 4);
            __bf16* d = &ldsA[row * 48 + c4 * 4];
            d[0] = (__bf16)av.x; d[1] = (__bf16)av.y;
            d[2] = (__bf16)av.z; d[3] = (__bf16)av.w;
        }
        // stage B tile transposed: 32(K) x 64(N) bf16, 8-byte loads (2 per thread)
        #pragma unroll
        for (int i = 0; i < 2; ++i) {
            int chunk = i * 256 + tid;          // 0..511, 4 bf16 each
            int kk = chunk >> 4;                // 16 chunks per K-row
            int n4 = (chunk & 15) * 4;
            short4 bv = *reinterpret_cast<const short4*>(
                Bw + (size_t)(k0 + kk) * N + tileN + n4);
            const short* bs = reinterpret_cast<const short*>(&bv);
            #pragma unroll
            for (int t = 0; t < 4; ++t)
                reinterpret_cast<short*>(ldsB)[(n4 + t) * 48 + kk] = bs[t];
        }
        // prefetch next A tile (global_prefetch_b8 on gfx1250)
        if (k0 + 32 < K) {
            int pr = tileM + (tid & 127);
            if (pr < M)
                __builtin_prefetch(A + (size_t)pr * K + k0 + 32, 0, 1);
        }
        __syncthreads();

        // Fragments per documented 16-bit layouts.
        // A 16x32: lanes 0-15 row=l K={0..7,16..23}; lanes 16-31 row=l K={8..15,24..31}
        FragB a0, a1, b0, b1;
        const __bf16* ar0 = &ldsA[(wm * 32 + l) * 48];
        const __bf16* ar1 = &ldsA[(wm * 32 + 16 + l) * 48];
        a0.i.lo = *reinterpret_cast<const int4*>(ar0 + hl * 8);
        a0.i.hi = *reinterpret_cast<const int4*>(ar0 + 16 + hl * 8);
        a1.i.lo = *reinterpret_cast<const int4*>(ar1 + hl * 8);
        a1.i.hi = *reinterpret_cast<const int4*>(ar1 + 16 + hl * 8);
        // B 32x16: lanes 0-15 col=l K={0..7,16..23}; lanes 16-31 col=l K={8..15,24..31}
        const __bf16* br0 = &ldsB[(wn * 32 + l) * 48];
        const __bf16* br1 = &ldsB[(wn * 32 + 16 + l) * 48];
        b0.i.lo = *reinterpret_cast<const int4*>(br0 + hl * 8);
        b0.i.hi = *reinterpret_cast<const int4*>(br0 + 16 + hl * 8);
        b1.i.lo = *reinterpret_cast<const int4*>(br1 + hl * 8);
        b1.i.hi = *reinterpret_cast<const int4*>(br1 + 16 + hl * 8);

        acc[0] = __builtin_amdgcn_wmma_f32_16x16x32_bf16(false, a0.v, false, b0.v,
                                                         (short)0, acc[0], false, false);
        acc[1] = __builtin_amdgcn_wmma_f32_16x16x32_bf16(false, a0.v, false, b1.v,
                                                         (short)0, acc[1], false, false);
        acc[2] = __builtin_amdgcn_wmma_f32_16x16x32_bf16(false, a1.v, false, b0.v,
                                                         (short)0, acc[2], false, false);
        acc[3] = __builtin_amdgcn_wmma_f32_16x16x32_bf16(false, a1.v, false, b1.v,
                                                         (short)0, acc[3], false, false);
    }

    // Epilogue.  C/D layout: VGPR j -> row = j + 8*hl (within 16x16 tile), col = l
    if (scatterOut) {
        #pragma unroll
        for (int mi = 0; mi < 2; ++mi) {
            #pragma unroll
            for (int j = 0; j < 8; ++j) {
                int rowg = tileM + wm * 32 + mi * 16 + j + hl * 8;
                if (rowg < M) {
                    int s = srcIdx[rowg];
                    int d = dstIdx[rowg];
                    #pragma unroll
                    for (int ni = 0; ni < 2; ++ni) {
                        int colg = tileN + wn * 32 + ni * 16 + l;
                        float v = acc[mi * 2 + ni][j] + bias[colg]
                                  + gatherTab[(size_t)s * N + colg];
                        v = fmaxf(v, 0.f);
                        atomicAdd(&scatterOut[(size_t)d * N + colg], v);
                    }
                }
            }
        }
    } else {
        #pragma unroll
        for (int mi = 0; mi < 2; ++mi) {
            #pragma unroll
            for (int j = 0; j < 8; ++j) {
                int rowg = tileM + wm * 32 + mi * 16 + j + hl * 8;
                if (rowg < M) {
                    #pragma unroll
                    for (int ni = 0; ni < 2; ++ni) {
                        int colg = tileN + wn * 32 + ni * 16 + l;
                        float v = acc[mi * 2 + ni][j] + bias[colg];
                        if (relu) v = fmaxf(v, 0.f);
                        Cout[(size_t)rowg * N + colg] = v;
                    }
                }
            }
        }
    }
}

// ---------------------------------------------------------------- attention
// one wave per (batch, head); S=32, hd=32
__launch_bounds__(32)
__global__ void attn_kernel(const float* __restrict__ Q, const float* __restrict__ K,
                            const float* __restrict__ V, float* __restrict__ O) {
    int bh = blockIdx.x;
    int b = bh / NHEAD, h = bh % NHEAD;
    size_t base = (size_t)b * S_DIM * C_DIM + h * HD;
    __shared__ float kt[S_DIM][HD];
    __shared__ float vt[S_DIM][HD];
    int lane = threadIdx.x;
    for (int i = 0; i < S_DIM; ++i) {
        kt[i][lane] = K[base + (size_t)i * C_DIM + lane];
        vt[i][lane] = V[base + (size_t)i * C_DIM + lane];
    }
    __syncthreads();
    float q[HD];
    for (int d = 0; d < HD; ++d) q[d] = Q[base + (size_t)lane * C_DIM + d];
    const float scale = 0.17677669529663687f;   // 1/sqrt(32)
    float sc[S_DIM];
    float mx = -1e30f;
    for (int j = 0; j < S_DIM; ++j) {
        float s = 0.f;
        for (int d = 0; d < HD; ++d) s += q[d] * kt[j][d];
        s *= scale;
        sc[j] = s;
        mx = fmaxf(mx, s);
    }
    float sum = 0.f;
    for (int j = 0; j < S_DIM; ++j) { sc[j] = __expf(sc[j] - mx); sum += sc[j]; }
    float inv = 1.f / sum;
    float out[HD];
    for (int d = 0; d < HD; ++d) out[d] = 0.f;
    for (int j = 0; j < S_DIM; ++j) {
        float a = sc[j] * inv;
        for (int d = 0; d < HD; ++d) out[d] += a * vt[j][d];
    }
    for (int d = 0; d < HD; ++d) O[base + (size_t)lane * C_DIM + d] = out[d];
}

// ---------------------------------------------------------------- layernorm (cols == 256 == blockDim)
__launch_bounds__(256)
__global__ void ln_kernel(const float* __restrict__ X, const float* __restrict__ R,
                          const float* __restrict__ w, const float* __restrict__ bvec,
                          float* __restrict__ Out) {
    int row = blockIdx.x;
    int tid = threadIdx.x;
    size_t idx = (size_t)row * C_DIM + tid;
    float v = X[idx] + (R ? R[idx] : 0.f);
    __shared__ float red[256];
    red[tid] = v; __syncthreads();
    for (int s = 128; s > 0; s >>= 1) { if (tid < s) red[tid] += red[tid + s]; __syncthreads(); }
    float mu = red[0] / (float)C_DIM; __syncthreads();
    float d = v - mu;
    red[tid] = d * d; __syncthreads();
    for (int s = 128; s > 0; s >>= 1) { if (tid < s) red[tid] += red[tid + s]; __syncthreads(); }
    float var = red[0] / (float)C_DIM;
    Out[idx] = d * rsqrtf(var + EPSV) * w[tid] + bvec[tid];
}

// ---------------------------------------------------------------- cls extraction
__global__ void cls_copy_kernel(const float* __restrict__ XT, float* __restrict__ cls) {
    int i = blockIdx.x * blockDim.x + threadIdx.x;
    if (i < B_DIM * C_DIM) {
        int b = i / C_DIM, c = i % C_DIM;
        cls[i] = XT[(size_t)b * S_DIM * C_DIM + c];          // token 0
    }
}

__launch_bounds__(256)
__global__ void cls_mean_kernel(const float* __restrict__ XT, float* __restrict__ cls_m) {
    int c = threadIdx.x;
    float s = 0.f;
    for (int b = 0; b < B_DIM; ++b)
        s += XT[(size_t)b * S_DIM * C_DIM + C_DIM + c];      // token 1
    cls_m[c] = s / (float)B_DIM;
}

// ---------------------------------------------------------------- batchnorm stats (LDS float atomics)
__launch_bounds__(256)
__global__ void bn_stats_kernel(const float* __restrict__ G, float* __restrict__ stats) {
    __shared__ float ps[HID];
    __shared__ float pq[HID];
    int tid = threadIdx.x;
    if (tid < HID) { ps[tid] = 0.f; pq[tid] = 0.f; }
    __syncthreads();
    size_t total = (size_t)N_NODES * HID;
    for (size_t idx = (size_t)blockIdx.x * blockDim.x + tid; idx < total;
         idx += (size_t)gridDim.x * blockDim.x) {
        float v = G[idx];
        int c = (int)(idx & (HID - 1));
        atomicAdd(&ps[c], v);
        atomicAdd(&pq[c], v * v);
    }
    __syncthreads();
    if (tid < HID) {
        atomicAdd(&stats[tid], ps[tid]);
        atomicAdd(&stats[HID + tid], pq[tid]);
    }
}

__global__ void bn_apply_kernel(const float* __restrict__ G2, const float* __restrict__ xg_in,
                                const float* __restrict__ stats,
                                const float* __restrict__ bw, const float* __restrict__ bb,
                                float* __restrict__ xg_out) {
    size_t idx = (size_t)blockIdx.x * blockDim.x + threadIdx.x;
    if (idx < (size_t)N_NODES * HID) {
        int c = (int)(idx & (HID - 1));
        float mu = stats[c] / (float)N_NODES;
        float var = stats[HID + c] / (float)N_NODES - mu * mu;
        float g = (G2[idx] - mu) * rsqrtf(var + EPSV) * bw[c] + bb[c];
        xg_out[idx] = (xg_in[idx] + fmaxf(g, 0.f)) * 0.5f;
    }
}

// ---------------------------------------------------------------- fusion MLP (FD=384, one block)
__launch_bounds__(FD)
__global__ void fusion_kernel(const float* __restrict__ cls_m, const float* __restrict__ xg0,
                              const float* __restrict__ lnf_w, const float* __restrict__ lnf_b,
                              const float* __restrict__ Wf1, const float* __restrict__ bf1,
                              const float* __restrict__ Wf2, const float* __restrict__ bf2,
                              const float* __restrict__ Wf3, const float* __restrict__ bf3,
                              float* __restrict__ xfin) {
    __shared__ float xv[FD], fv[FD], tv[FD], red[512];
    int tid = threadIdx.x;
    float xval = (tid < C_DIM) ? cls_m[tid] : xg0[tid - C_DIM];
    xv[tid] = xval;
    red[tid] = xval; if (tid < 128) red[FD + tid] = 0.f;
    __syncthreads();
    for (int s = 256; s > 0; s >>= 1) { if (tid < s) red[tid] += red[tid + s]; __syncthreads(); }
    float mu = red[0] / (float)FD; __syncthreads();
    float d = xval - mu;
    red[tid] = d * d; if (tid < 128) red[FD + tid] = 0.f;
    __syncthreads();
    for (int s = 256; s > 0; s >>= 1) { if (tid < s) red[tid] += red[tid + s]; __syncthreads(); }
    float var = red[0] / (float)FD;
    fv[tid] = d * rsqrtf(var + EPSV) * lnf_w[tid] + lnf_b[tid];
    __syncthreads();

    float a = bf1[tid];
    for (int i = 0; i < FD; ++i) a += fv[i] * Wf1[(size_t)i * FD + tid];
    a = (a > 0.f) ? a : 0.01f * a;
    tv[tid] = a; __syncthreads();

    float b2 = bf2[tid];
    for (int i = 0; i < FD; ++i) b2 += tv[i] * Wf2[(size_t)i * FD + tid];
    b2 = (b2 > 0.f) ? b2 : 0.01f * b2;
    __syncthreads(); fv[tid] = b2; __syncthreads();

    float c3 = bf3[tid];
    for (int i = 0; i < FD; ++i) c3 += fv[i] * Wf3[(size_t)i * FD + tid];

    xfin[tid] = (xv[tid] + c3) * 0.5f;
}

// ---------------------------------------------------------------- outputs
__global__ void out_tab_kernel(const float* __restrict__ XT, const float* __restrict__ cls,
                               const float* __restrict__ xfin, float* __restrict__ out) {
    size_t idx = (size_t)blockIdx.x * blockDim.x + threadIdx.x;
    size_t total = (size_t)B_DIM * S_DIM * C_DIM;
    if (idx < total) {
        int c = (int)(idx % C_DIM);
        size_t r = idx / C_DIM;
        int s = (int)(r % S_DIM);
        size_t b = r / S_DIM;
        out[idx] = (s == 0) ? (xfin[c] + 0.5f * cls[b * C_DIM + c]) : XT[idx];
    }
}

__global__ void out_gnn_kernel(const float* __restrict__ XG, const float* __restrict__ xfin,
                               float* __restrict__ out) {
    size_t idx = (size_t)blockIdx.x * blockDim.x + threadIdx.x;
    size_t total = (size_t)N_NODES * HID;
    if (idx < total) {
        int c = (int)(idx & (HID - 1));
        size_t i = idx >> 7;
        out[idx] = (i == 0) ? (XG[c] + 0.5f * xfin[C_DIM + c]) : XG[idx];
    }
}

// ---------------------------------------------------------------- driver

extern "C" void kernel_launch(void* const* d_in, const int* in_sizes, int n_in,
                              void* d_out, int out_size, void* d_ws, size_t ws_size,
                              hipStream_t stream) {
    const float* x_tab     = (const float*)d_in[0];
    const float* x_gnn     = (const float*)d_in[1];
    const float* edge_attr = (const float*)d_in[2];
    const int*   edge_idx  = (const int*)d_in[3];
    const float* Wq = (const float*)d_in[4];  const float* bq = (const float*)d_in[5];
    const float* Wk = (const float*)d_in[6];  const float* bk = (const float*)d_in[7];
    const float* Wv = (const float*)d_in[8];  const float* bv = (const float*)d_in[9];
    const float* Wo = (const float*)d_in[10]; const float* bo = (const float*)d_in[11];
    const float* ln1_w = (const float*)d_in[12]; const float* ln1_b = (const float*)d_in[13];
    const float* W1 = (const float*)d_in[14]; const float* b1 = (const float*)d_in[15];
    const float* W2 = (const float*)d_in[16]; const float* b2 = (const float*)d_in[17];
    const float* ln2_w = (const float*)d_in[18]; const float* ln2_b = (const float*)d_in[19];
    const float* tabn_w = (const float*)d_in[20]; const float* tabn_b = (const float*)d_in[21];
    const float* We = (const float*)d_in[22]; const float* be = (const float*)d_in[23];
    const float* Wm1 = (const float*)d_in[24]; const float* bm1 = (const float*)d_in[25];
    const float* Wm2 = (const float*)d_in[26]; const float* bm2 = (const float*)d_in[27];
    const float* bn_w = (const float*)d_in[28]; const float* bn_b = (const float*)d_in[29];
    const float* lnf_w = (const float*)d_in[30]; const float* lnf_b = (const float*)d_in[31];
    const float* Wf1 = (const float*)d_in[32]; const float* bf1 = (const float*)d_in[33];
    const float* Wf2 = (const float*)d_in[34]; const float* bf2 = (const float*)d_in[35];
    const float* Wf3 = (const float*)d_in[36]; const float* bf3 = (const float*)d_in[37];

    char* ws = (char*)d_ws;
    size_t off = 0;
    auto take = [&](size_t bytes) -> char* {
        char* p = ws + off;
        off += (bytes + 255) & ~(size_t)255;
        return p;
    };

    __bf16* Wq_bf  = (__bf16*)take((size_t)C_DIM * C_DIM * 2);
    __bf16* Wk_bf  = (__bf16*)take((size_t)C_DIM * C_DIM * 2);
    __bf16* Wv_bf  = (__bf16*)take((size_t)C_DIM * C_DIM * 2);
    __bf16* Wo_bf  = (__bf16*)take((size_t)C_DIM * C_DIM * 2);
    __bf16* W1_bf  = (__bf16*)take((size_t)C_DIM * C_DIM * 2);
    __bf16* W2_bf  = (__bf16*)take((size_t)C_DIM * C_DIM * 2);
    __bf16* We_bf  = (__bf16*)take((size_t)HID * HID * 2);
    __bf16* Wm1_bf = (__bf16*)take((size_t)HID * HID * 2);
    __bf16* Wm2_bf = (__bf16*)take((size_t)HID * HID * 2);

    const size_t TAB = (size_t)B_DIM * S_DIM * C_DIM;    // 8,388,608
    const size_t GN  = (size_t)N_NODES * HID;            // 6,400,000
    float* bufQ = (float*)take(TAB * 4);
    float* bufK = (float*)take(TAB * 4);
    float* bufV = (float*)take(TAB * 4);
    float* bufA = (float*)take(TAB * 4);
    float* agg   = (float*)take(GN * 4);
    float* gbuf  = (float*)take(GN * 4);
    float* tbuf  = (float*)take(GN * 4);
    float* xgbuf = (float*)take(GN * 4);
    float* cls   = (float*)take((size_t)B_DIM * C_DIM * 4);
    float* cls_m = (float*)take((size_t)C_DIM * 4);
    float* stats = (float*)take((size_t)2 * HID * 4);
    float* xfin  = (float*)take((size_t)FD * 4);

    // --- weights -> bf16 ---
    auto cvt = [&](const float* s, __bf16* d, int n) {
        cvt_bf16_kernel<<<(n + 255) / 256, 256, 0, stream>>>(s, d, n);
    };
    cvt(Wq, Wq_bf, C_DIM * C_DIM);  cvt(Wk, Wk_bf, C_DIM * C_DIM);
    cvt(Wv, Wv_bf, C_DIM * C_DIM);  cvt(Wo, Wo_bf, C_DIM * C_DIM);
    cvt(W1, W1_bf, C_DIM * C_DIM);  cvt(W2, W2_bf, C_DIM * C_DIM);
    cvt(We, We_bf, HID * HID);      cvt(Wm1, Wm1_bf, HID * HID);
    cvt(Wm2, Wm2_bf, HID * HID);

    // --- transformer ---
    const int BS = B_DIM * S_DIM;
    dim3 gTab(BS / 128, C_DIM / 64);         // (256, 4)
    gemm_wmma_kernel<<<gTab, 256, 0, stream>>>(x_tab, Wq_bf, bq, bufQ, BS, C_DIM, C_DIM, 0,
                                               nullptr, nullptr, nullptr, nullptr);
    gemm_wmma_kernel<<<gTab, 256, 0, stream>>>(x_tab, Wk_bf, bk, bufK, BS, C_DIM, C_DIM, 0,
                                               nullptr, nullptr, nullptr, nullptr);
    gemm_wmma_kernel<<<gTab, 256, 0, stream>>>(x_tab, Wv_bf, bv, bufV, BS, C_DIM, C_DIM, 0,
                                               nullptr, nullptr, nullptr, nullptr);
    attn_kernel<<<B_DIM * NHEAD, 32, 0, stream>>>(bufQ, bufK, bufV, bufA);
    gemm_wmma_kernel<<<gTab, 256, 0, stream>>>(bufA, Wo_bf, bo, bufK, BS, C_DIM, C_DIM, 0,
                                               nullptr, nullptr, nullptr, nullptr);
    ln_kernel<<<BS, 256, 0, stream>>>(bufK, x_tab, ln1_w, ln1_b, bufQ);           // h
    gemm_wmma_kernel<<<gTab, 256, 0, stream>>>(bufQ, W1_bf, b1, bufV, BS, C_DIM, C_DIM, 1,
                                               nullptr, nullptr, nullptr, nullptr);
    gemm_wmma_kernel<<<gTab, 256, 0, stream>>>(bufV, W2_bf, b2, bufK, BS, C_DIM, C_DIM, 0,
                                               nullptr, nullptr, nullptr, nullptr);
    ln_kernel<<<BS, 256, 0, stream>>>(bufK, bufQ, ln2_w, ln2_b, bufA);            // h2
    ln_kernel<<<BS, 256, 0, stream>>>(bufA, nullptr, tabn_w, tabn_b, bufK);       // xt (full)
    cls_copy_kernel<<<(B_DIM * C_DIM + 255) / 256, 256, 0, stream>>>(bufK, cls);
    cls_mean_kernel<<<1, 256, 0, stream>>>(bufK, cls_m);

    // --- GNN ---
    zero_kernel<<<(int)((GN + 255) / 256), 256, 0, stream>>>(agg, GN);
    zero_kernel<<<1, 256, 0, stream>>>(stats, 2 * HID);
    dim3 gE(E_EDGES / 128, HID / 64);        // (5000, 2)  fused gather+GEMM+relu+scatter
    gemm_wmma_kernel<<<gE, 256, 0, stream>>>(edge_attr, We_bf, be, nullptr,
                                             E_EDGES, HID, HID, 0,
                                             edge_idx, edge_idx + E_EDGES, x_gnn, agg);
    add_kernel<<<(int)((GN + 255) / 256), 256, 0, stream>>>(x_gnn, agg, gbuf, GN);
    dim3 gG((N_NODES + 127) / 128, HID / 64);   // (391, 2)
    gemm_wmma_kernel<<<gG, 256, 0, stream>>>(gbuf, Wm1_bf, bm1, tbuf, N_NODES, HID, HID, 1,
                                             nullptr, nullptr, nullptr, nullptr);
    gemm_wmma_kernel<<<gG, 256, 0, stream>>>(tbuf, Wm2_bf, bm2, agg, N_NODES, HID, HID, 0,
                                             nullptr, nullptr, nullptr, nullptr);
    bn_stats_kernel<<<512, 256, 0, stream>>>(agg, stats);
    bn_apply_kernel<<<(int)((GN + 255) / 256), 256, 0, stream>>>(agg, x_gnn, stats,
                                                                 bn_w, bn_b, xgbuf);

    // --- fusion + outputs ---
    fusion_kernel<<<1, FD, 0, stream>>>(cls_m, xgbuf, lnf_w, lnf_b,
                                        Wf1, bf1, Wf2, bf2, Wf3, bf3, xfin);
    out_tab_kernel<<<(int)((TAB + 255) / 256), 256, 0, stream>>>(bufK, cls, xfin,
                                                                 (float*)d_out);
    out_gnn_kernel<<<(int)((GN + 255) / 256), 256, 0, stream>>>(xgbuf, xfin,
                                                                (float*)d_out + TAB);
}
